// SelfAttention_16071767621785
// MI455X (gfx1250) — matrix operand order repaired and checked by reading, not compiled, not run on hardware
//
#include <hip/hip_runtime.h>

typedef __attribute__((ext_vector_type(16))) __bf16 v16bf;
typedef __attribute__((ext_vector_type(8)))  __bf16 v8bf;
typedef __attribute__((ext_vector_type(8)))  float  v8f;

#define N_EMBD 1024
#define N_HEAD 16
#define HEADD  64
#define BATCH  4
#define SEQ    2048
#define ROWS   (BATCH*SEQ)   /* 8192 */

static __device__ __forceinline__ __bf16 f2bf(float x) { return (__bf16)x; }

// 16 contiguous bf16 -> v16bf (two 16B loads, zero-cost shuffle)
static __device__ __forceinline__ v16bf ld16c(const __bf16* p) {
    v8bf lo = *(const v8bf*)p;
    v8bf hi = *(const v8bf*)(p + 8);
    return __builtin_shufflevector(lo, hi, 0,1,2,3,4,5,6,7,8,9,10,11,12,13,14,15);
}
// B-fragment gather: elements {p[0..7], p[16..23]}
static __device__ __forceinline__ v16bf ld16gap(const __bf16* p) {
    v8bf lo = *(const v8bf*)p;
    v8bf hi = *(const v8bf*)(p + 16);
    return __builtin_shufflevector(lo, hi, 0,1,2,3,4,5,6,7,8,9,10,11,12,13,14,15);
}
static __device__ __forceinline__ v8f wmma_bf16(v16bf a, v16bf b, v8f c) {
    return __builtin_amdgcn_wmma_f32_16x16x32_bf16(false, a, false, b, (short)0, c,
                                                   false, false);
}
// all-reduce max across each 16-lane half via DPP16 (masks {1,2,7,15} generate
// the full group; max is commutative+idempotent so the reduction is exact)
static __device__ __forceinline__ float dpp_max16(float v) {
    int x;
    x = __builtin_amdgcn_update_dpp(0, __float_as_int(v), 0xB1,  0xF, 0xF, true);
    v = fmaxf(v, __int_as_float(x));   // quad_perm(1,0,3,2)  : xor 1
    x = __builtin_amdgcn_update_dpp(0, __float_as_int(v), 0x4E,  0xF, 0xF, true);
    v = fmaxf(v, __int_as_float(x));   // quad_perm(2,3,0,1)  : xor 2
    x = __builtin_amdgcn_update_dpp(0, __float_as_int(v), 0x141, 0xF, 0xF, true);
    v = fmaxf(v, __int_as_float(x));   // row_half_mirror     : xor 7
    x = __builtin_amdgcn_update_dpp(0, __float_as_int(v), 0x140, 0xF, 0xF, true);
    v = fmaxf(v, __int_as_float(x));   // row_mirror          : xor 15
    return v;
}

// ---------------------------------------------------------------------------
// x (f32) -> bf16
// ---------------------------------------------------------------------------
__global__ __launch_bounds__(256)
void cvt_bf16(const float* __restrict__ x, __bf16* __restrict__ y)
{
    const size_t i = ((size_t)blockIdx.x * 256 + threadIdx.x) * 8;
    v8bf o;
    #pragma unroll
    for (int j = 0; j < 8; ++j) o[j] = f2bf(x[i + j]);
    *(v8bf*)(y + i) = o;
}

// ---------------------------------------------------------------------------
// Transpose+convert the four 1024x1024 weight matrices: Wt[n][k] = bf16(W[k][n])
// ---------------------------------------------------------------------------
__global__ __launch_bounds__(256)
void transpose_w4(const float* __restrict__ Wq, const float* __restrict__ Wk,
                  const float* __restrict__ Wv, const float* __restrict__ Wp,
                  __bf16* __restrict__ Tq, __bf16* __restrict__ Tk,
                  __bf16* __restrict__ Tv, __bf16* __restrict__ Tp)
{
    __shared__ __bf16 L[64 * 68];
    const float* W; __bf16* T;
    switch (blockIdx.z) {
        case 0:  W = Wq; T = Tq; break;
        case 1:  W = Wk; T = Tk; break;
        case 2:  W = Wv; T = Tv; break;
        default: W = Wp; T = Tp; break;
    }
    const int k0 = blockIdx.y * 64, n0 = blockIdx.x * 64;
    const int tid = threadIdx.x;
    {
        const int n = tid & 63, kq = tid >> 6;
        #pragma unroll
        for (int i = 0; i < 16; ++i) {
            const int k = kq * 16 + i;
            L[n * 68 + k] = f2bf(W[(size_t)(k0 + k) * N_EMBD + n0 + n]);
        }
    }
    __syncthreads();
    {
        const int k = tid & 63, nq = tid >> 6;
        #pragma unroll
        for (int i = 0; i < 16; ++i) {
            const int n2 = nq * 16 + i;
            T[(size_t)(n0 + n2) * N_EMBD + k0 + k] = L[n2 * 68 + k];
        }
    }
}

// ---------------------------------------------------------------------------
// GEMM: C[M,N] = (A[M,K]*Bt[N,K]^T + bias) * scale.  A,Bt bf16.
// MODE 0: bf16 row-major out; MODE 1: bf16 transposed-per-head (Vt[B*H*Dh][T],
// 8 t's per 16B store); MODE 2: f32 row-major out.
// Block 256 (8 waves), 128x64 tile; wave = 32x32 (4 WMMA tiles: better
// FLOP per ds_load_b128 than 16x32). LDS fragment-major: every fragment is
// 2x ds_load_b128, every staging write one 16B ds_store.
// ---------------------------------------------------------------------------
template<int MODE>
__global__ __launch_bounds__(256)
void gemm_bias_wmma(const __bf16* __restrict__ A,
                    const __bf16* __restrict__ Bt,
                    const float* __restrict__ bias,
                    void* __restrict__ Cout,
                    float scale, int M, int N, int K)
{
    __shared__ __bf16 As[128 * 40];   // [row][perm-K(32)] pad->40
    __shared__ __bf16 Bs[64 * 40];    // [col][K(32)]      pad->40

    const int tid  = threadIdx.x;
    const int lane = tid & 31;
    const int w    = tid >> 5;
    const int m0   = blockIdx.y * 128;
    const int n0   = blockIdx.x * 64;
    const int wm   = (w & 3) * 32;
    const int wn   = (w >> 2) * 32;
    const int lrow = lane & 15;
    const int lhi  = lane >> 4;

    v8f acc[2][2];
    #pragma unroll
    for (int i = 0; i < 2; ++i)
        #pragma unroll
        for (int j = 0; j < 2; ++j) {
            v8f z = {0.f,0.f,0.f,0.f,0.f,0.f,0.f,0.f};
            acc[i][j] = z;
        }

    // A staging: 128x32 = 16 elems/thread (two 8-blocks); perm {0->0,8->16,16->8,24->24}
    const int arow  = tid >> 1;
    const int akoff = (tid & 1) * 16;                       // 0 or 16
    const int apb0  = (akoff == 0) ? 0  : 8;                // perm(akoff)
    const int apb1  = (akoff == 0) ? 16 : 24;               // perm(akoff+8)
    // B staging: 64x32 = 8 elems/thread
    const int scol  = tid >> 2;
    const int bkoff = (tid & 3) * 8;

    for (int kk = 0; kk < K; kk += 32) {
        const uint4* asrc = (const uint4*)(A + (size_t)(m0 + arow) * K + kk + akoff);
        *(uint4*)&As[arow * 40 + apb0] = asrc[0];
        *(uint4*)&As[arow * 40 + apb1] = asrc[1];
        *(uint4*)&Bs[scol * 40 + bkoff] =
            *(const uint4*)(Bt + (size_t)(n0 + scol) * K + kk + bkoff);
        __syncthreads();

        const v16bf a0 = ld16c(&As[(wm + lrow)      * 40 + lhi * 16]);
        const v16bf a1 = ld16c(&As[(wm + 16 + lrow) * 40 + lhi * 16]);
        const v16bf b0 = ld16gap(&Bs[(wn + lrow)      * 40 + lhi * 8]);
        const v16bf b1 = ld16gap(&Bs[(wn + 16 + lrow) * 40 + lhi * 8]);
        acc[0][0] = wmma_bf16(a0, b0, acc[0][0]);
        acc[0][1] = wmma_bf16(a0, b1, acc[0][1]);
        acc[1][0] = wmma_bf16(a1, b0, acc[1][0]);
        acc[1][1] = wmma_bf16(a1, b1, acc[1][1]);
        __syncthreads();
    }

    #pragma unroll
    for (int mt = 0; mt < 2; ++mt)
        #pragma unroll
        for (int nt = 0; nt < 2; ++nt) {
            const int col  = n0 + wn + nt * 16 + lrow;
            const int row0 = m0 + wm + mt * 16 + lhi * 8;
            const float bia = bias[col];
            if (MODE == 1) {
                v8bf pk;
                #pragma unroll
                for (int r = 0; r < 8; ++r) pk[r] = f2bf((acc[mt][nt][r] + bia) * scale);
                __bf16* Vt = (__bf16*)Cout;
                const int bb = row0 >> 11, tt = row0 & 2047;
                *(v8bf*)(Vt + (size_t)(bb * 1024 + col) * SEQ + tt) = pk;
            } else if (MODE == 0) {
                __bf16* Cb = (__bf16*)Cout;
                #pragma unroll
                for (int r = 0; r < 8; ++r)
                    Cb[(size_t)(row0 + r) * N + col] = f2bf((acc[mt][nt][r] + bia) * scale);
            } else {
                float* Cf = (float*)Cout;
                #pragma unroll
                for (int r = 0; r < 8; ++r)
                    Cf[(size_t)(row0 + r) * N + col] = (acc[mt][nt][r] + bia) * scale;
            }
        }
}

// ---------------------------------------------------------------------------
// One 32-key block of causal flash attention. MASKED only for the diagonal
// block; all strictly-below-diagonal blocks skip the compares entirely.
// Row sums come from an extra P x ones WMMA (accL); row max via DPP16.
// ---------------------------------------------------------------------------
template<bool MASKED>
static __device__ __forceinline__ void attn_kblock(
    const int key0, const int qbase, const int lrow, const int lhi,
    const __bf16* __restrict__ Kp,   // Kb + b,h row base
    const __bf16* __restrict__ Vp,   // Vt + b,h row base
    const v16bf qa0, const v16bf qa1, const v16bf ones,
    __bf16* __restrict__ Pw, float* __restrict__ m,
    v8f* __restrict__ accO, v8f& accL)
{
    const int akb = lhi * 8;
    v8f s0 = {0.f,0.f,0.f,0.f,0.f,0.f,0.f,0.f};
    v8f s1 = {0.f,0.f,0.f,0.f,0.f,0.f,0.f,0.f};
    {
        const __bf16* k0p = Kp + (size_t)(key0 + lrow) * N_EMBD + lhi * 16;
        const __bf16* k1p = k0p + (size_t)16 * N_EMBD;
        s0 = wmma_bf16(qa0, ld16c(k0p),      s0);
        s0 = wmma_bf16(qa1, ld16c(k0p + 32), s0);
        s1 = wmma_bf16(qa0, ld16c(k1p),      s1);
        s1 = wmma_bf16(qa1, ld16c(k1p + 32), s1);
    }

    #pragma unroll
    for (int r = 0; r < 8; ++r) {
        float t0 = s0[r], t1 = s1[r];
        if (MASKED) {
            const int qg = qbase + lhi * 8 + r;
            t0 = (key0 + lrow      <= qg) ? t0 : -1e30f;
            t1 = (key0 + 16 + lrow <= qg) ? t1 : -1e30f;
        }
        const float blk   = dpp_max16(fmaxf(t0, t1));
        const float mnew  = fmaxf(m[r], blk);
        const float alpha = __expf(m[r] - mnew);
        m[r] = mnew;
        #pragma unroll
        for (int c = 0; c < 4; ++c) accO[c][r] *= alpha;
        accL[r] *= alpha;
        const int Mloc = lhi * 8 + r;
        Pw[Mloc * 40 + lrow]      = f2bf(__expf(t0 - mnew));  // LDS in-order in-wave
        Pw[Mloc * 40 + 16 + lrow] = f2bf(__expf(t1 - mnew));
    }

    const v16bf pa = ld16c(Pw + lrow * 40 + lhi * 16);
    accL = wmma_bf16(pa, ones, accL);                  // row sums for free
    #pragma unroll
    for (int c = 0; c < 4; ++c) {
        const __bf16* vp = Vp + (size_t)(c * 16 + lrow) * SEQ + key0 + lhi * 16;
        accO[c] = wmma_bf16(pa, ld16c(vp), accO[c]);
    }
}

// ---------------------------------------------------------------------------
// Causal flash attention. 256 threads = 8 INDEPENDENT waves (no barriers),
// 16 query rows per wave; K/Vt fragments stream from global (K+V bf16 = 32MB,
// L2 = 192MB -> L2-resident). Q is pre-scaled by 1/sqrt(Dh) in its GEMM.
// ---------------------------------------------------------------------------
__global__ __launch_bounds__(256)
void flash_attn_wmma(const __bf16* __restrict__ Qb,
                     const __bf16* __restrict__ Kb,
                     const __bf16* __restrict__ Vt,
                     __bf16* __restrict__ Ob)
{
    __shared__ __bf16 Ps[8 * 16 * 40];

    const int tid  = threadIdx.x;
    const int lane = tid & 31;
    const int w    = tid >> 5;
    const int lrow = lane & 15;
    const int lhi  = lane >> 4;

    const int bh = blockIdx.y;
    const int b  = bh >> 4;
    const int h  = bh & 15;
    const int qbase = blockIdx.x * 128 + w * 16;

    const __bf16* qrow = Qb + ((size_t)(b * SEQ + qbase + lrow)) * N_EMBD + h * HEADD;
    const v16bf qa0 = ld16c(qrow + lhi * 16);
    const v16bf qa1 = ld16c(qrow + 32 + lhi * 16);

    v16bf ones;
    #pragma unroll
    for (int i = 0; i < 16; ++i) ones[i] = f2bf(1.0f);

    float m[8];
    #pragma unroll
    for (int r = 0; r < 8; ++r) m[r] = -1e30f;
    v8f accO[4];
    #pragma unroll
    for (int c = 0; c < 4; ++c) {
        v8f z = {0.f,0.f,0.f,0.f,0.f,0.f,0.f,0.f};
        accO[c] = z;
    }
    v8f accL = {0.f,0.f,0.f,0.f,0.f,0.f,0.f,0.f};

    const int nkb = (qbase >> 5) + 1;
    __bf16* Pw = Ps + w * (16 * 40);
    const __bf16* Kp = Kb + (size_t)(b * SEQ) * N_EMBD + h * HEADD;
    const __bf16* Vp = Vt + (size_t)(b * 1024 + h * HEADD) * SEQ;

    for (int kbi = 0; kbi < nkb - 1; ++kbi) {           // strictly-below-diagonal
        const int key0 = kbi * 32;
        __builtin_prefetch(Kp + (size_t)(key0 + 32 + lane) * N_EMBD, 0, 1);
        __builtin_prefetch(Vp + (size_t)lane        * SEQ + key0 + 32, 0, 1);
        __builtin_prefetch(Vp + (size_t)(lane + 32) * SEQ + key0 + 32, 0, 1);
        attn_kblock<false>(key0, qbase, lrow, lhi, Kp, Vp, qa0, qa1, ones,
                           Pw, m, accO, accL);
    }
    attn_kblock<true>((nkb - 1) * 32, qbase, lrow, lhi, Kp, Vp, qa0, qa1, ones,
                      Pw, m, accO, accL);               // diagonal block

    #pragma unroll
    for (int r = 0; r < 8; ++r) {
        const float inv = 1.0f / accL[r];
        const int rowg = qbase + lhi * 8 + r;
        __bf16* orow = Ob + ((size_t)(b * SEQ + rowg)) * N_EMBD + h * HEADD;
        #pragma unroll
        for (int c = 0; c < 4; ++c)
            orow[c * 16 + lrow] = f2bf(accO[c][r] * inv);
    }
}

// ---------------------------------------------------------------------------
extern "C" void kernel_launch(void* const* d_in, const int* in_sizes, int n_in,
                              void* d_out, int out_size, void* d_ws, size_t ws_size,
                              hipStream_t stream)
{
    const float* x  = (const float*)d_in[0];
    const float* Wq = (const float*)d_in[1];
    const float* bq = (const float*)d_in[2];
    const float* Wk = (const float*)d_in[3];
    const float* bk = (const float*)d_in[4];
    const float* Wv = (const float*)d_in[5];
    const float* bv = (const float*)d_in[6];
    const float* Wp = (const float*)d_in[7];
    const float* bp = (const float*)d_in[8];

    const size_t EM = (size_t)ROWS * N_EMBD;         // 8M elems
    const size_t WM = (size_t)N_EMBD * N_EMBD;       // 1M elems
    __bf16* xb = (__bf16*)d_ws;                      // 16MB (reused as Obuf)
    __bf16* Tq = xb + EM;
    __bf16* Tk = Tq + WM;
    __bf16* Tv = Tk + WM;
    __bf16* Tp = Tv + WM;
    __bf16* Qb = Tp + WM;
    __bf16* Kb = Qb + EM;
    __bf16* Vt = Kb + EM;                            // total 72MB
    __bf16* Obuf = xb;   // safe: x fully consumed before attention writes O

    cvt_bf16<<<dim3(EM / (256 * 8)), 256, 0, stream>>>(x, xb);
    transpose_w4<<<dim3(16, 16, 4), 256, 0, stream>>>(Wq, Wk, Wv, Wp, Tq, Tk, Tv, Tp);

    const float qscale = 0.125f;                     // 1/sqrt(64) folded into Q
    dim3 gg(N_EMBD / 64, ROWS / 128);                // (16, 64)
    gemm_bias_wmma<0><<<gg, 256, 0, stream>>>(xb, Tq, bq, Qb, qscale, ROWS, N_EMBD, N_EMBD);
    gemm_bias_wmma<0><<<gg, 256, 0, stream>>>(xb, Tk, bk, Kb, 1.0f,   ROWS, N_EMBD, N_EMBD);
    gemm_bias_wmma<1><<<gg, 256, 0, stream>>>(xb, Tv, bv, Vt, 1.0f,   ROWS, N_EMBD, N_EMBD);

    flash_attn_wmma<<<dim3(SEQ / 128, BATCH * N_HEAD), 256, 0, stream>>>(Qb, Kb, Vt, Obuf);

    gemm_bias_wmma<2><<<gg, 256, 0, stream>>>(Obuf, Tp, bp, d_out, 1.0f, ROWS, N_EMBD, N_EMBD);
}
